// DecoderBlock_37873021616206
// MI455X (gfx1250) — matrix operand order repaired and checked by reading, not compiled, not run on hardware
//
#include <hip/hip_runtime.h>
#include <math.h>

// ---------------- problem constants ----------------
constexpr int Bc  = 4;
constexpr int S   = 2048;      // S_TGT == S_SRC
constexpr int Dm  = 512;
constexpr int Hh  = 8;
constexpr int DKc = 64;        // Dm / Hh
constexpr int FFc = 2048;
constexpr int Mrows = Bc * S;  // 8192
#define EPSLN 1e-6f

// ---------------- types ----------------
typedef __bf16        bf16_t;
typedef bf16_t        v16bf  __attribute__((ext_vector_type(16)));
typedef float         v8f    __attribute__((ext_vector_type(8)));
typedef unsigned int  u32x4  __attribute__((ext_vector_type(4)));
typedef unsigned int  u32x4v __attribute__((ext_vector_type(4)));
typedef int           i32x4v __attribute__((ext_vector_type(4)));
typedef int           i32x8v __attribute__((ext_vector_type(8)));

union Frag16 {                       // 16 bf16 elements == 8 VGPRs
    unsigned short u[16];
    u32x4          q[2];
    v16bf          v;
};

__device__ __forceinline__ unsigned short f32_to_bf16(float f) {
    unsigned int u = __float_as_uint(f);
    u += 0x7FFFu + ((u >> 16) & 1u);          // round-to-nearest-even
    return (unsigned short)(u >> 16);
}

__device__ __forceinline__ v8f zero8() {
    v8f z;
#pragma unroll
    for (int i = 0; i < 8; ++i) z[i] = 0.f;
    return z;
}

// LDS byte-offset of a __shared__ object (generic -> addrspace(3) -> u32).
// Cast via u64 first so the host pass (64-bit pointers) also type-checks.
typedef __attribute__((address_space(3))) const void* lds_cvptr;
__device__ __forceinline__ unsigned lds_off(const void* p) {
    return (unsigned)(unsigned long long)(lds_cvptr)p;
}

// ---------------- Tensor Data Mover: 2-D bf16 tile -> LDS ----------------
// D# per CDNA5 ISA ch.8: group0 = {count/type/lds_addr/global_addr},
// group1 = {data_size, tensor dims, tile dims, dim0 stride}.
__device__ __forceinline__ void tdm_load_2d_bf16(const unsigned short* gsrc,
                                                 unsigned ldsByteAddr,
                                                 int tileW /*elems*/,
                                                 int tileH /*rows*/,
                                                 int strideElems,
                                                 int tensorD0, int tensorD1) {
    unsigned long long ga = (unsigned long long)gsrc;
    u32x4v g0;
    g0[0] = 1u;                                        // count=1, user mode
    g0[1] = ldsByteAddr;                               // bits 63:32 lds_addr
    g0[2] = (unsigned)ga;                              // global_addr lo
    g0[3] = ((unsigned)(ga >> 32) & 0x01FFFFFFu)       // global_addr hi (57b)
            | (2u << 30);                              // type = 2 ("image")
    i32x8v g1;
    g1[0] = (int)(1u << 16);                           // data_size=1 (2 bytes)
    g1[1] = (int)(((unsigned)tensorD0 & 0xFFFFu) << 16);          // td0[15:0]
    g1[2] = (int)((((unsigned)tensorD0 >> 16) & 0xFFFFu) |
                  (((unsigned)tensorD1 & 0xFFFFu) << 16));        // td0 hi | td1 lo
    g1[3] = (int)((((unsigned)tensorD1 >> 16) & 0xFFFFu) |
                  (((unsigned)tileW & 0xFFFFu) << 16));           // td1 hi | tile_dim0
    g1[4] = (int)((unsigned)tileH & 0xFFFFu);          // tile_dim1 (tile_dim2=0)
    g1[5] = strideElems;                               // tensor_dim0_stride[31:0]
    g1[6] = 0;                                         // stride hi / dim1 stride
    g1[7] = 0;
    i32x4v z4 = {0, 0, 0, 0};
#if defined(__clang_major__) && (__clang_major__ >= 23)
    i32x8v z8 = {0, 0, 0, 0, 0, 0, 0, 0};
    __builtin_amdgcn_tensor_load_to_lds(g0, g1, z4, z4, z8, 0);
#else
    __builtin_amdgcn_tensor_load_to_lds(g0, g1, z4, z4, 0);
#endif
}

// ---------------- elementwise conversion kernels ----------------
__global__ void cvt_f32_bf16_kernel(const float* __restrict__ in,
                                    unsigned short* __restrict__ out, size_t n) {
    size_t i = (size_t)blockIdx.x * blockDim.x + threadIdx.x;
    if (i < n) out[i] = f32_to_bf16(in[i]);
}

// W[K,N] f32 -> Wt[N,K] bf16  (so each WMMA B-fragment lane-load is contiguous)
__global__ void wt_cvt_kernel(const float* __restrict__ W,
                              unsigned short* __restrict__ Wt, int K, int N) {
    size_t i = (size_t)blockIdx.x * blockDim.x + threadIdx.x;
    if (i >= (size_t)N * K) return;
    int k = (int)(i % K);
    int n = (int)(i / K);
    Wt[i] = f32_to_bf16(W[(size_t)k * N + n]);
}

// in [B,S,H*DK] bf16 -> out [B,H,S,DK] bf16
__global__ void split_heads_kernel(const unsigned short* __restrict__ in,
                                   unsigned short* __restrict__ out) {
    size_t i = (size_t)blockIdx.x * blockDim.x + threadIdx.x;
    if (i >= (size_t)Mrows * Dm) return;
    int d = (int)(i % DKc); size_t r = i / DKc;
    int s = (int)(r % S);   size_t r2 = r / S;
    int h = (int)(r2 % Hh); int b = (int)(r2 / Hh);
    out[i] = in[((size_t)b * S + s) * Dm + h * DKc + d];
}

// in [B,S,H*DK] bf16 -> out Vt [B,H,DK,S] bf16 (transposed for PV B-fragments)
__global__ void transpose_v_kernel(const unsigned short* __restrict__ in,
                                   unsigned short* __restrict__ out) {
    size_t i = (size_t)blockIdx.x * blockDim.x + threadIdx.x;
    if (i >= (size_t)Mrows * Dm) return;
    int s = (int)(i % S);   size_t r = i / S;
    int d = (int)(r % DKc); size_t r2 = r / DKc;
    int h = (int)(r2 % Hh); int b = (int)(r2 / Hh);
    out[i] = in[((size_t)b * S + s) * Dm + h * DKc + d];
}

// ---------------- layernorm (ddof=1, eps outside sqrt) -> bf16 ----------------
__global__ __launch_bounds__(256)
void layernorm_kernel(const float* __restrict__ x,
                      const float* __restrict__ g,
                      const float* __restrict__ beta,
                      unsigned short* __restrict__ out, int D) {
    __shared__ float red[256];
    const int row = blockIdx.x;
    const float* xr = x + (size_t)row * D;

    float s = 0.f;
    for (int i = threadIdx.x; i < D; i += blockDim.x) s += xr[i];
    red[threadIdx.x] = s; __syncthreads();
    for (int o = 128; o > 0; o >>= 1) {
        if ((int)threadIdx.x < o) red[threadIdx.x] += red[threadIdx.x + o];
        __syncthreads();
    }
    const float mean = red[0] / (float)D;
    __syncthreads();

    float v = 0.f;
    for (int i = threadIdx.x; i < D; i += blockDim.x) {
        float d = xr[i] - mean; v += d * d;
    }
    red[threadIdx.x] = v; __syncthreads();
    for (int o = 128; o > 0; o >>= 1) {
        if ((int)threadIdx.x < o) red[threadIdx.x] += red[threadIdx.x + o];
        __syncthreads();
    }
    const float stdv = sqrtf(red[0] / (float)(D - 1));
    const float inv  = 1.f / (stdv + EPSLN);

    for (int i = threadIdx.x; i < D; i += blockDim.x) {
        float y = g[i] * (xr[i] - mean) * inv + beta[i];
        out[(size_t)row * D + i] = f32_to_bf16(y);
    }
}

// ---------------- WMMA GEMM with TDM-staged A and pipelined fragments --------
// Grid: (M/16, N/512). Block = 256 threads = 8 waves; all waves share one
// 16-row A strip (TDM double-buffered into LDS in 16x256 chunks); wave w
// computes columns n0 = blockIdx.y*512 + w*64 (4 accumulators).
template <bool ADD_BIAS, bool ADD_RES, bool RELU, bool OUT_BF16>
__global__ __launch_bounds__(256)
void gemm_wmma_kernel(const unsigned short* __restrict__ A,
                      const unsigned short* __restrict__ Wt,
                      const float* __restrict__ bias,
                      const float* __restrict__ res,
                      float* __restrict__ outF,
                      unsigned short* __restrict__ outB,
                      int M, int N, int K) {
    constexpr int KC = 256;                       // A chunk width (elements)
    constexpr int KSTEPS = KC / 32;
    __shared__ unsigned short Atile[2][16 * KC];  // double-buffered A strip

    const int lane = threadIdx.x & 31;
    const int lrow = lane & 15;
    const int hi   = lane >> 4;
    const int wave = threadIdx.x >> 5;
    const int m0   = blockIdx.x << 4;
    const int n0   = (blockIdx.y << 9) + (wave << 6);

    const int nchunks = K / KC;
    const unsigned short* Ablk = A + (size_t)m0 * K;

    if (wave == 0) {
        tdm_load_2d_bf16(Ablk, lds_off(&Atile[0][0]), KC, 16, K, K, M - m0);
        __builtin_amdgcn_s_wait_tensorcnt(0);
    }
    __syncthreads();

    v8f acc[4];
#pragma unroll
    for (int t = 0; t < 4; ++t) acc[t] = zero8();

    const unsigned short* Bbase[4];
#pragma unroll
    for (int t = 0; t < 4; ++t)
        Bbase[t] = Wt + (size_t)(n0 + t * 16 + lrow) * K + (hi ? 16 : 0);

    for (int c = 0; c < nchunks; ++c) {
        if (wave == 0 && (c + 1) < nchunks)       // kick next DMA while computing
            tdm_load_2d_bf16(Ablk + (c + 1) * KC, lds_off(&Atile[(c + 1) & 1][0]),
                             KC, 16, K, K, M - m0);

        const unsigned short* Arow = &Atile[c & 1][0] + lrow * KC + (hi ? 16 : 0);
        const int cbase = c * KC;

        Frag16 aF[2];
        Frag16 bF[2][4];
        // preload step 0
        aF[0].q[0] = *reinterpret_cast<const u32x4*>(Arow);
        aF[0].q[1] = *reinterpret_cast<const u32x4*>(Arow + 8);
#pragma unroll
        for (int t = 0; t < 4; ++t) {
            const unsigned short* bp = Bbase[t] + cbase;
            bF[0][t].q[0] = *reinterpret_cast<const u32x4*>(bp);
            bF[0][t].q[1] = *reinterpret_cast<const u32x4*>(bp + 8);
        }
#pragma unroll
        for (int s = 0; s < KSTEPS; ++s) {
            const int cur = s & 1, nxt = cur ^ 1;
            if (s + 1 < KSTEPS) {                 // prefetch next k-step fragments
                const unsigned short* ap = Arow + (s + 1) * 32;
                aF[nxt].q[0] = *reinterpret_cast<const u32x4*>(ap);
                aF[nxt].q[1] = *reinterpret_cast<const u32x4*>(ap + 8);
                const int kg = cbase + (s + 1) * 32;
#pragma unroll
                for (int t = 0; t < 4; ++t) {
                    const unsigned short* bp = Bbase[t] + kg;
                    bF[nxt][t].q[0] = *reinterpret_cast<const u32x4*>(bp);
                    bF[nxt][t].q[1] = *reinterpret_cast<const u32x4*>(bp + 8);
                }
            }
#pragma unroll
            for (int t = 0; t < 4; ++t)
                acc[t] = __builtin_amdgcn_wmma_f32_16x16x32_bf16(
                    false, aF[cur].v, false, bF[cur][t].v, (short)0, acc[t],
                    false, false);
        }

        if (wave == 0 && (c + 1) < nchunks)
            __builtin_amdgcn_s_wait_tensorcnt(0);
        __syncthreads();
    }

    const int rowbase = m0 + (hi << 3);           // C: lane=col, 8 rows per half
#pragma unroll
    for (int t = 0; t < 4; ++t) {
        const int col = n0 + t * 16 + lrow;
        const float bv = ADD_BIAS ? bias[col] : 0.f;
#pragma unroll
        for (int v = 0; v < 8; ++v) {
            const int row = rowbase + v;
            float val = acc[t][v] + bv;
            if (RELU) val = fmaxf(val, 0.f);
            const size_t idx = (size_t)row * N + col;
            if (ADD_RES) val += res[idx];
            if (OUT_BF16) outB[idx] = f32_to_bf16(val);
            else          outF[idx] = val;
        }
    }
}

// ---------------- flash-style attention, one wave per (b,h, 16 q-rows) ----------
// Qh,Kh: [B,H,S,64] bf16 ; Vt: [B,H,64,S] bf16 ; O: [B,S,512] bf16 (heads merged)
template <bool CAUSAL>
__global__ __launch_bounds__(32)
void attn_kernel(const unsigned short* __restrict__ Qh,
                 const unsigned short* __restrict__ Kh,
                 const unsigned short* __restrict__ Vt,
                 unsigned short* __restrict__ O) {
    __shared__ unsigned short Pl[16 * 32];       // P tile staged for C->A reshape
    const int lane = threadIdx.x & 31;
    const int lrow = lane & 15;
    const int hi   = lane >> 4;
    const int q0   = blockIdx.x * 16;
    const int bh   = blockIdx.y;                 // b*H + h
    const int b    = bh >> 3, h = bh & 7;

    const size_t baseQ = (size_t)bh * S * DKc;
    const size_t baseK = (size_t)bh * S * DKc;
    const size_t baseV = (size_t)bh * DKc * S;

    Frag16 qa0, qa1;
    {
        const unsigned short* qr = Qh + baseQ + (size_t)(q0 + lrow) * DKc + (hi ? 16 : 0);
        qa0.q[0] = *reinterpret_cast<const u32x4*>(qr);
        qa0.q[1] = *reinterpret_cast<const u32x4*>(qr + 8);
        qa1.q[0] = *reinterpret_cast<const u32x4*>(qr + 32);
        qa1.q[1] = *reinterpret_cast<const u32x4*>(qr + 40);
    }

    v8f acc[4];
    float m_i[8], l_i[8];
#pragma unroll
    for (int t = 0; t < 4; ++t) acc[t] = zero8();
#pragma unroll
    for (int v = 0; v < 8; ++v) { m_i[v] = -1e30f; l_i[v] = 0.f; }

    const int rb   = q0 + (hi << 3);
    const int jend = CAUSAL ? (q0 + 16) : S;

    for (int j0 = 0; j0 < jend; j0 += 32) {
        v8f s0 = zero8(), s1 = zero8();
        {
            const unsigned short* kr =
                Kh + baseK + (size_t)(j0 + lrow) * DKc + (hi ? 16 : 0);
            Frag16 kb;
            kb.q[0] = *reinterpret_cast<const u32x4*>(kr);
            kb.q[1] = *reinterpret_cast<const u32x4*>(kr + 8);
            s0 = __builtin_amdgcn_wmma_f32_16x16x32_bf16(false, qa0.v, false, kb.v,
                                                         (short)0, s0, false, false);
            kb.q[0] = *reinterpret_cast<const u32x4*>(kr + 32);
            kb.q[1] = *reinterpret_cast<const u32x4*>(kr + 40);
            s0 = __builtin_amdgcn_wmma_f32_16x16x32_bf16(false, qa1.v, false, kb.v,
                                                         (short)0, s0, false, false);
        }
        {
            const unsigned short* kr =
                Kh + baseK + (size_t)(j0 + 16 + lrow) * DKc + (hi ? 16 : 0);
            Frag16 kb;
            kb.q[0] = *reinterpret_cast<const u32x4*>(kr);
            kb.q[1] = *reinterpret_cast<const u32x4*>(kr + 8);
            s1 = __builtin_amdgcn_wmma_f32_16x16x32_bf16(false, qa0.v, false, kb.v,
                                                         (short)0, s1, false, false);
            kb.q[0] = *reinterpret_cast<const u32x4*>(kr + 32);
            kb.q[1] = *reinterpret_cast<const u32x4*>(kr + 40);
            s1 = __builtin_amdgcn_wmma_f32_16x16x32_bf16(false, qa1.v, false, kb.v,
                                                         (short)0, s1, false, false);
        }

        const int c0 = j0 + lrow, c1 = j0 + 16 + lrow;
#pragma unroll
        for (int v = 0; v < 8; ++v) {
            float a   = s0[v] * 0.125f;          // 1/sqrt(64)
            float bsc = s1[v] * 0.125f;
            if (CAUSAL) {
                if (c0 > rb + v) a   = -1e9f;
                if (c1 > rb + v) bsc = -1e9f;
            }
            float mx = fmaxf(a, bsc);
#pragma unroll
            for (int off = 1; off < 16; off <<= 1)
                mx = fmaxf(mx, __shfl_xor(mx, off, 32));
            const float mnew  = fmaxf(m_i[v], mx);
            const float alpha = __expf(m_i[v] - mnew);
            const float e0 = __expf(a - mnew), e1 = __expf(bsc - mnew);
            float rs = e0 + e1;
#pragma unroll
            for (int off = 1; off < 16; off <<= 1)
                rs += __shfl_xor(rs, off, 32);
            l_i[v] = l_i[v] * alpha + rs;
            m_i[v] = mnew;
#pragma unroll
            for (int t = 0; t < 4; ++t) acc[t][v] *= alpha;
            Pl[(rb - q0 + v) * 32 + lrow]      = f32_to_bf16(e0);
            Pl[(rb - q0 + v) * 32 + 16 + lrow] = f32_to_bf16(e1);
        }
        __syncthreads();

        Frag16 pa;
        {
            const unsigned short* pr = &Pl[lrow * 32 + (hi ? 16 : 0)];
            pa.q[0] = *reinterpret_cast<const u32x4*>(pr);
            pa.q[1] = *reinterpret_cast<const u32x4*>(pr + 8);
        }
#pragma unroll
        for (int t = 0; t < 4; ++t) {
            const unsigned short* vr =
                Vt + baseV + (size_t)(t * 16 + lrow) * S + j0 + (hi ? 16 : 0);
            Frag16 vb;
            vb.q[0] = *reinterpret_cast<const u32x4*>(vr);
            vb.q[1] = *reinterpret_cast<const u32x4*>(vr + 8);
            acc[t] = __builtin_amdgcn_wmma_f32_16x16x32_bf16(
                false, pa.v, false, vb.v, (short)0, acc[t], false, false);
        }
        __syncthreads();
    }

#pragma unroll
    for (int t = 0; t < 4; ++t) {
#pragma unroll
        for (int v = 0; v < 8; ++v) {
            const int qrow = rb + v;
            const float val = acc[t][v] / l_i[v];
            O[((size_t)b * S + qrow) * Dm + h * DKc + t * 16 + lrow] =
                f32_to_bf16(val);
        }
    }
}

// ---------------- host orchestration ----------------
static inline int ceil_div(size_t a, size_t b) { return (int)((a + b - 1) / b); }

extern "C" void kernel_launch(void* const* d_in, const int* in_sizes, int n_in,
                              void* d_out, int out_size, void* d_ws, size_t ws_size,
                              hipStream_t stream) {
    (void)in_sizes; (void)n_in; (void)out_size; (void)ws_size;

    const float* x     = (const float*)d_in[0];
    const float* enc   = (const float*)d_in[1];
    const float* sa_wq = (const float*)d_in[4];
    const float* sa_wk = (const float*)d_in[5];
    const float* sa_wv = (const float*)d_in[6];
    const float* sa_wo = (const float*)d_in[7];
    const float* sa_bo = (const float*)d_in[8];
    const float* ca_wq = (const float*)d_in[9];
    const float* ca_wk = (const float*)d_in[10];
    const float* ca_wv = (const float*)d_in[11];
    const float* ca_wo = (const float*)d_in[12];
    const float* ca_bo = (const float*)d_in[13];
    const float* ff_w1 = (const float*)d_in[14];
    const float* ff_b1 = (const float*)d_in[15];
    const float* ff_w2 = (const float*)d_in[16];
    const float* ff_b2 = (const float*)d_in[17];
    const float* ln0_g = (const float*)d_in[18];
    const float* ln0_b = (const float*)d_in[19];
    const float* ln1_g = (const float*)d_in[20];
    const float* ln1_b = (const float*)d_in[21];
    const float* ln2_g = (const float*)d_in[22];
    const float* ln2_b = (const float*)d_in[23];
    float* xout = (float*)d_out;

    char* p = (char*)d_ws;
    auto alloc = [&](size_t elems) -> unsigned short* {
        unsigned short* r = (unsigned short*)p;
        p += ((elems * sizeof(unsigned short)) + 255) & ~(size_t)255;
        return r;
    };

    unsigned short* hbf    = alloc((size_t)Mrows * Dm);
    unsigned short* encbf  = alloc((size_t)Mrows * Dm);
    unsigned short* qp     = alloc((size_t)Mrows * Dm);
    unsigned short* kp     = alloc((size_t)Mrows * Dm);
    unsigned short* vp     = alloc((size_t)Mrows * Dm);
    unsigned short* Qhd    = alloc((size_t)Mrows * Dm);
    unsigned short* Khd    = alloc((size_t)Mrows * Dm);
    unsigned short* Vtr    = alloc((size_t)Mrows * Dm);
    unsigned short* attnO  = alloc((size_t)Mrows * Dm);
    unsigned short* ff1    = alloc((size_t)Mrows * FFc);
    unsigned short* sa_wq_t = alloc((size_t)Dm * Dm);
    unsigned short* sa_wk_t = alloc((size_t)Dm * Dm);
    unsigned short* sa_wv_t = alloc((size_t)Dm * Dm);
    unsigned short* sa_wo_t = alloc((size_t)Dm * Dm);
    unsigned short* ca_wq_t = alloc((size_t)Dm * Dm);
    unsigned short* ca_wk_t = alloc((size_t)Dm * Dm);
    unsigned short* ca_wv_t = alloc((size_t)Dm * Dm);
    unsigned short* ca_wo_t = alloc((size_t)Dm * Dm);
    unsigned short* ff_w1_t = alloc((size_t)Dm * FFc);
    unsigned short* ff_w2_t = alloc((size_t)FFc * Dm);

    const int EW = 256;
    const size_t nMD = (size_t)Mrows * Dm;

    wt_cvt_kernel<<<ceil_div((size_t)Dm * Dm, EW), EW, 0, stream>>>(sa_wq, sa_wq_t, Dm, Dm);
    wt_cvt_kernel<<<ceil_div((size_t)Dm * Dm, EW), EW, 0, stream>>>(sa_wk, sa_wk_t, Dm, Dm);
    wt_cvt_kernel<<<ceil_div((size_t)Dm * Dm, EW), EW, 0, stream>>>(sa_wv, sa_wv_t, Dm, Dm);
    wt_cvt_kernel<<<ceil_div((size_t)Dm * Dm, EW), EW, 0, stream>>>(sa_wo, sa_wo_t, Dm, Dm);
    wt_cvt_kernel<<<ceil_div((size_t)Dm * Dm, EW), EW, 0, stream>>>(ca_wq, ca_wq_t, Dm, Dm);
    wt_cvt_kernel<<<ceil_div((size_t)Dm * Dm, EW), EW, 0, stream>>>(ca_wk, ca_wk_t, Dm, Dm);
    wt_cvt_kernel<<<ceil_div((size_t)Dm * Dm, EW), EW, 0, stream>>>(ca_wv, ca_wv_t, Dm, Dm);
    wt_cvt_kernel<<<ceil_div((size_t)Dm * Dm, EW), EW, 0, stream>>>(ca_wo, ca_wo_t, Dm, Dm);
    wt_cvt_kernel<<<ceil_div((size_t)Dm * FFc, EW), EW, 0, stream>>>(ff_w1, ff_w1_t, Dm, FFc);
    wt_cvt_kernel<<<ceil_div((size_t)FFc * Dm, EW), EW, 0, stream>>>(ff_w2, ff_w2_t, FFc, Dm);
    cvt_f32_bf16_kernel<<<ceil_div(nMD, EW), EW, 0, stream>>>(enc, encbf, nMD);

    auto gemm_grid = [](int M, int N) { return dim3(M >> 4, N >> 9); };
    const dim3 attn_grid(S / 16, Bc * Hh);

    // ================= self-attention block =================
    layernorm_kernel<<<Mrows, 256, 0, stream>>>(x, ln0_g, ln0_b, hbf, Dm);
    gemm_wmma_kernel<false, false, false, true><<<gemm_grid(Mrows, Dm), 256, 0, stream>>>(
        hbf, sa_wq_t, nullptr, nullptr, nullptr, qp, Mrows, Dm, Dm);
    gemm_wmma_kernel<false, false, false, true><<<gemm_grid(Mrows, Dm), 256, 0, stream>>>(
        hbf, sa_wk_t, nullptr, nullptr, nullptr, kp, Mrows, Dm, Dm);
    gemm_wmma_kernel<false, false, false, true><<<gemm_grid(Mrows, Dm), 256, 0, stream>>>(
        hbf, sa_wv_t, nullptr, nullptr, nullptr, vp, Mrows, Dm, Dm);
    split_heads_kernel<<<ceil_div(nMD, EW), EW, 0, stream>>>(qp, Qhd);
    split_heads_kernel<<<ceil_div(nMD, EW), EW, 0, stream>>>(kp, Khd);
    transpose_v_kernel<<<ceil_div(nMD, EW), EW, 0, stream>>>(vp, Vtr);
    attn_kernel<true><<<attn_grid, 32, 0, stream>>>(Qhd, Khd, Vtr, attnO);
    gemm_wmma_kernel<true, true, false, false><<<gemm_grid(Mrows, Dm), 256, 0, stream>>>(
        attnO, sa_wo_t, sa_bo, x, xout, nullptr, Mrows, Dm, Dm);

    // ================= cross-attention block =================
    layernorm_kernel<<<Mrows, 256, 0, stream>>>(xout, ln1_g, ln1_b, hbf, Dm);
    gemm_wmma_kernel<false, false, false, true><<<gemm_grid(Mrows, Dm), 256, 0, stream>>>(
        hbf, ca_wq_t, nullptr, nullptr, nullptr, qp, Mrows, Dm, Dm);
    gemm_wmma_kernel<false, false, false, true><<<gemm_grid(Mrows, Dm), 256, 0, stream>>>(
        encbf, ca_wk_t, nullptr, nullptr, nullptr, kp, Mrows, Dm, Dm);
    gemm_wmma_kernel<false, false, false, true><<<gemm_grid(Mrows, Dm), 256, 0, stream>>>(
        encbf, ca_wv_t, nullptr, nullptr, nullptr, vp, Mrows, Dm, Dm);
    split_heads_kernel<<<ceil_div(nMD, EW), EW, 0, stream>>>(qp, Qhd);
    split_heads_kernel<<<ceil_div(nMD, EW), EW, 0, stream>>>(kp, Khd);
    transpose_v_kernel<<<ceil_div(nMD, EW), EW, 0, stream>>>(vp, Vtr);
    attn_kernel<false><<<attn_grid, 32, 0, stream>>>(Qhd, Khd, Vtr, attnO);
    gemm_wmma_kernel<true, true, false, false><<<gemm_grid(Mrows, Dm), 256, 0, stream>>>(
        attnO, ca_wo_t, ca_bo, xout, xout, nullptr, Mrows, Dm, Dm);

    // ================= feed-forward block =================
    layernorm_kernel<<<Mrows, 256, 0, stream>>>(xout, ln2_g, ln2_b, hbf, Dm);
    gemm_wmma_kernel<true, false, true, true><<<gemm_grid(Mrows, FFc), 256, 0, stream>>>(
        hbf, ff_w1_t, ff_b1, nullptr, nullptr, ff1, Mrows, FFc, Dm);
    gemm_wmma_kernel<true, true, false, false><<<gemm_grid(Mrows, Dm), 256, 0, stream>>>(
        ff1, ff_w2_t, ff_b2, xout, xout, nullptr, Mrows, Dm, FFc);
}
